// Mutual_Attention_89532888252583
// MI455X (gfx1250) — matrix operand order repaired and checked by reading, not compiled, not run on hardware
//
#include <hip/hip_runtime.h>

// Flash-attention forward for MI455X (gfx1250, wave32, WMMA bf16 16x16x32).
// out = softmax(q k^T / sqrt(512)) @ v  over N=8192 tokens, d=512.
// sum_of_means == 1/N analytically (softmax rows sum to 1).

#define N_TOK  8192
#define DMODEL 512
#define BM     64      // query rows per block
#define BN     32      // key rows per tile
#define BLOCK  256     // 8 wave32 waves

typedef __attribute__((ext_vector_type(16))) __bf16       v16bf;
typedef __attribute__((ext_vector_type(8)))  float        v8f;
typedef __attribute__((ext_vector_type(4)))  unsigned int u32x4;

struct U32x8 { u32x4 lo; u32x4 hi; };

// float -> bf16 round-to-nearest-even
__device__ __forceinline__ unsigned short f2bf(float x) {
  unsigned int u = __builtin_bit_cast(unsigned int, x);
  u += 0x7fffu + ((u >> 16) & 1u);
  return (unsigned short)(u >> 16);
}

// A-operand (16x32 bf16): lanes 0-15 take K 0..15, lanes 16-31 take K 16..31 (contiguous)
__device__ __forceinline__ v16bf load_a(const unsigned short* p, int lh) {
  U32x8 r;
  r.lo = *(const u32x4*)(p + 16 * lh);
  r.hi = *(const u32x4*)(p + 16 * lh + 8);
  return __builtin_bit_cast(v16bf, r);
}
// B-operand (32x16 bf16): per lane 16 contiguous K values (caller adds 16*lh)
__device__ __forceinline__ v16bf load_b(const unsigned short* p) {
  U32x8 r;
  r.lo = *(const u32x4*)(p);
  r.hi = *(const u32x4*)(p + 8);
  return __builtin_bit_cast(v16bf, r);
}

// XOR lane shuffle via ds_swizzle (group-of-32: and=0x1F, or=0, xor=m)
#define XOR_SWIZ(x, m)                                                        \
  __builtin_bit_cast(float, __builtin_amdgcn_ds_swizzle(                      \
      __builtin_bit_cast(int, (x)), (0x1F | ((m) << 10))))

// LDS layout (bytes)
#define SQ_OFF    0
#define SK_OFF    (BM * DMODEL * 2)             // 65536
#define SVT_OFF   (SK_OFF + BN * DMODEL * 2)    // +32768
#define SP_OFF    (SVT_OFF + DMODEL * BN * 2)   // +32768
#define SMAX_OFF  (SP_OFF + BM * BN * 2)        // +4096
#define SSUM_OFF  (SMAX_OFF + 2 * BM * 4)       // +512
#define SMEM_BYTES (SSUM_OFF + 2 * BM * 4)      // 136192 total

__global__ __launch_bounds__(BLOCK, 1)
void fa_fwd(const float* __restrict__ Q, const float* __restrict__ K,
            const float* __restrict__ V, float* __restrict__ Out) {
  extern __shared__ char smem[];
  unsigned short* sQ  = (unsigned short*)(smem + SQ_OFF);   // [BM][DMODEL] bf16, pre-scaled
  unsigned short* sK  = (unsigned short*)(smem + SK_OFF);   // [BN][DMODEL] bf16
  unsigned short* sVt = (unsigned short*)(smem + SVT_OFF);  // [DMODEL][BN] bf16 (transposed)
  unsigned short* sP  = (unsigned short*)(smem + SP_OFF);   // [BM][BN]     bf16
  float* sMax = (float*)(smem + SMAX_OFF);                  // [2][BM] partial row max
  float* sSum = (float*)(smem + SSUM_OFF);                  // [2][BM] partial row sum

  const int tid   = threadIdx.x;
  const int lane  = tid & 31;
  const int wave  = tid >> 5;
  const int strip = wave >> 1;   // 0..3 : which 16-row query strip
  const int nw    = wave & 1;    // 0..1 : which 256-col half of V / 16-col half of S
  const int lh    = lane >> 4;   // lane half
  const int l16   = lane & 15;

  const int m0 = blockIdx.x * BM;

  // ---- load Q tile, fold softmax scale * log2(e) so we can use v_exp_f32 ----
  const float qscale = 0.04419417382415922f * 1.4426950408889634f;
  {
    const float4* qg = (const float4*)(Q + (size_t)m0 * DMODEL);
    for (int i = tid; i < BM * DMODEL / 4; i += BLOCK) {
      float4 qv = qg[i];
      int o = i * 4;
      sQ[o + 0] = f2bf(qv.x * qscale);
      sQ[o + 1] = f2bf(qv.y * qscale);
      sQ[o + 2] = f2bf(qv.z * qscale);
      sQ[o + 3] = f2bf(qv.w * qscale);
    }
  }

  const v8f vzero = {0.f, 0.f, 0.f, 0.f, 0.f, 0.f, 0.f, 0.f};
  v8f acc[16];                                   // O strip: 16 rows x 256 cols (16 tiles)
#pragma unroll
  for (int t = 0; t < 16; ++t) acc[t] = vzero;

  float mrun[8], lrun[8];                        // per-lane row = strip*16 + 8*lh + d
#pragma unroll
  for (int d = 0; d < 8; ++d) { mrun[d] = -__builtin_inff(); lrun[d] = 0.f; }

  const unsigned short* qrow = sQ + (strip * 16 + l16) * DMODEL;  // A rows: M = l16
  const unsigned short* prow = sP + (strip * 16 + l16) * BN;

  for (int n0 = 0; n0 < N_TOK; n0 += BN) {
    __syncthreads();  // previous iteration finished with sK/sVt/sP/stats

    // ---- cooperative tile load: K row-major, V transposed, both bf16 ----
    {
      const float4* kg = (const float4*)(K + (size_t)n0 * DMODEL);
      const float4* vg = (const float4*)(V + (size_t)n0 * DMODEL);
      for (int i = tid; i < BN * DMODEL / 4; i += BLOCK) {
        float4 kv = kg[i];
        int o = i * 4;
        sK[o + 0] = f2bf(kv.x); sK[o + 1] = f2bf(kv.y);
        sK[o + 2] = f2bf(kv.z); sK[o + 3] = f2bf(kv.w);
        float4 vv = vg[i];
        int r = o >> 9;          // o / DMODEL
        int c = o & (DMODEL - 1);
        sVt[(c + 0) * BN + r] = f2bf(vv.x);
        sVt[(c + 1) * BN + r] = f2bf(vv.y);
        sVt[(c + 2) * BN + r] = f2bf(vv.z);
        sVt[(c + 3) * BN + r] = f2bf(vv.w);
      }
    }
    // prefetch next K/V tile (64 KB each, 256 B per thread) into cache
    int n1 = n0 + BN;
    if (n1 < N_TOK) {
      __builtin_prefetch((const char*)(K + (size_t)n1 * DMODEL) + tid * 256, 0, 1);
      __builtin_prefetch((const char*)(V + (size_t)n1 * DMODEL) + tid * 256, 0, 1);
    }
    __syncthreads();

    // ---- S(16x16) = Qstrip * K^T for this wave's 16 key columns ----
    v8f s = vzero;
    const unsigned short* krow = sK + (nw * 16 + l16) * DMODEL;  // B cols: N = l16
#pragma unroll
    for (int kk = 0; kk < DMODEL; kk += 32) {
      v16bf a = load_a(qrow + kk, lh);
      v16bf b = load_b(krow + kk + 16 * lh);
      s = __builtin_amdgcn_wmma_f32_16x16x32_bf16(false, a, false, b,
                                                  (short)0, s, false, false);
    }

    // ---- partial row max over this 16-col slice (XOR-reduce 16 lanes) ----
    float tmax[8];
#pragma unroll
    for (int d = 0; d < 8; ++d) {
      float x = s[d];
      x = fmaxf(x, XOR_SWIZ(x, 1));
      x = fmaxf(x, XOR_SWIZ(x, 2));
      x = fmaxf(x, XOR_SWIZ(x, 4));
      x = fmaxf(x, XOR_SWIZ(x, 8));
      tmax[d] = x;
    }
    if (l16 == 0) {  // lanes 0 (rows +0..7) and 16 (rows +8..15)
#pragma unroll
      for (int d = 0; d < 8; ++d)
        sMax[nw * BM + strip * 16 + 8 * lh + d] = tmax[d];
    }
    __syncthreads();

    // ---- online softmax: combine maxes, exponentiate, partial sums, emit P ----
    float alpha[8], tsum[8];
#pragma unroll
    for (int d = 0; d < 8; ++d) {
      int r = strip * 16 + 8 * lh + d;
      float tm = fmaxf(sMax[r], sMax[BM + r]);
      float mn = fmaxf(mrun[d], tm);
      alpha[d] = __builtin_amdgcn_exp2f(mrun[d] - mn);  // exp2(-inf)=0 on first tile
      mrun[d]  = mn;
      float p  = __builtin_amdgcn_exp2f(s[d] - mn);
      float x  = p;
      x += XOR_SWIZ(x, 1);
      x += XOR_SWIZ(x, 2);
      x += XOR_SWIZ(x, 4);
      x += XOR_SWIZ(x, 8);
      tsum[d] = x;
      sP[r * BN + (nw * 16 + l16)] = f2bf(p);  // C-layout -> row-major LDS
    }
    if (l16 == 0) {
#pragma unroll
      for (int d = 0; d < 8; ++d)
        sSum[nw * BM + strip * 16 + 8 * lh + d] = tsum[d];
    }
    __syncthreads();

    // ---- l update + rescale O by alpha ----
#pragma unroll
    for (int d = 0; d < 8; ++d) {
      int r = strip * 16 + 8 * lh + d;
      lrun[d] = lrun[d] * alpha[d] + (sSum[r] + sSum[BM + r]);
    }
#pragma unroll
    for (int t = 0; t < 16; ++t)
#pragma unroll
      for (int d = 0; d < 8; ++d) acc[t][d] *= alpha[d];

    // ---- O += P(16x32) @ V(32x256half) : 16 WMMAs, A reused across tiles ----
    v16bf pa = load_a(prow, lh);
#pragma unroll
    for (int t = 0; t < 16; ++t) {
      const unsigned short* vb = sVt + (nw * 256 + t * 16 + l16) * BN + 16 * lh;
      v16bf b = load_b(vb);
      acc[t] = __builtin_amdgcn_wmma_f32_16x16x32_bf16(false, pa, false, b,
                                                       (short)0, acc[t], false, false);
    }
  }

  // ---- epilogue: O / l, store fp32 ----
#pragma unroll
  for (int d = 0; d < 8; ++d) lrun[d] = 1.0f / lrun[d];
#pragma unroll
  for (int t = 0; t < 16; ++t) {
    int col = nw * 256 + t * 16 + l16;
#pragma unroll
    for (int d = 0; d < 8; ++d) {
      int row = m0 + strip * 16 + 8 * lh + d;
      Out[(size_t)row * DMODEL + col] = acc[t][d] * lrun[d];
    }
  }
  // sum_of_means: softmax rows sum to 1 -> mean of row means == 1/N
  if (blockIdx.x == 0 && tid == 0)
    Out[(size_t)N_TOK * DMODEL] = 1.0f / (float)N_TOK;
}

extern "C" void kernel_launch(void* const* d_in, const int* in_sizes, int n_in,
                              void* d_out, int out_size, void* d_ws, size_t ws_size,
                              hipStream_t stream) {
  (void)in_sizes; (void)n_in; (void)out_size; (void)d_ws; (void)ws_size;
  const float* q = (const float*)d_in[0];  // q_token [N, 512]
  const float* k = (const float*)d_in[1];  // k_token [N, 512]
  const float* v = (const float*)d_in[2];  // v_token [N, 512]
  // d_in[3..5] (Wq, Wk, Wv) are dead in the reference output.
  float* out = (float*)d_out;              // [N*512] out ++ [1] sum_of_means

  dim3 grid(N_TOK / BM);
  fa_fwd<<<grid, BLOCK, SMEM_BYTES, stream>>>(q, k, v, out);
}
